// BNN_FASHION_CNN_20143396618240
// MI455X (gfx1250) — 8-bit matrix operand order repaired and checked by reading, compiled, not run on hardware
//
#include <hip/hip_runtime.h>

typedef int v8i __attribute__((ext_vector_type(8)));
typedef int v4i __attribute__((ext_vector_type(4)));

#define EPSV 1e-5f

__device__ __forceinline__ int iclamp(int v, int lo, int hi) {
    return v < lo ? lo : (v > hi ? hi : v);
}

__global__ void pack_signs(const float* __restrict__ fc1_w,
                           const float* __restrict__ fc2_w,
                           signed char* __restrict__ w1s,
                           signed char* __restrict__ w2s) {
    const long long n1 = 2048LL * 3136LL;
    const long long n2 = 10LL * 2048LL;
    long long stride = (long long)gridDim.x * blockDim.x;
    for (long long i = (long long)blockIdx.x * blockDim.x + threadIdx.x;
         i < n1 + n2; i += stride) {
        if (i < n1) {
            w1s[i] = (fc1_w[i] >= 0.f) ? (signed char)1 : (signed char)-1;
        } else {
            long long j = i - n1;
            w2s[j] = (fc2_w[j] >= 0.f) ? (signed char)1 : (signed char)-1;
        }
    }
}

__global__ void conv1_k(const float* __restrict__ x,
                        const float* __restrict__ conv1_w,
                        const float* __restrict__ g1, const float* __restrict__ bt1,
                        const float* __restrict__ mu1, const float* __restrict__ vr1,
                        signed char* __restrict__ p1) {
    __shared__ float sX[784];
    __shared__ float sW[576];
    __shared__ float sInv[64], sMu[64], sBt[64];

    const int b = blockIdx.x;
    const int tid = threadIdx.x;

    for (int i = tid; i < 784; i += 256) sX[i] = x[(size_t)b * 784 + i];
    for (int i = tid; i < 576; i += 256) sW[i] = (conv1_w[i] >= 0.f) ? 1.f : -1.f;
    for (int c = tid; c < 64; c += 256) {
        sInv[c] = g1[c] * rsqrtf(vr1[c] + EPSV);
        sMu[c] = mu1[c];
        sBt[c] = bt1[c];
    }
    __syncthreads();

    for (int item = tid; item < 64 * 196; item += 256) {
        const int c  = item / 196;
        const int pp = item % 196;
        const int py = pp / 14, px = pp % 14;
        bool any = false;
        #pragma unroll
        for (int sy = 0; sy < 2; ++sy) {
            #pragma unroll
            for (int sx = 0; sx < 2; ++sx) {
                const int oy = 2 * py + sy, ox = 2 * px + sx;
                float h = 0.f;
                #pragma unroll
                for (int dy = 0; dy < 3; ++dy) {
                    const int iy = iclamp(oy + dy - 1, 0, 27);
                    #pragma unroll
                    for (int dx = 0; dx < 3; ++dx) {
                        const int ix = iclamp(ox + dx - 1, 0, 27);
                        h += sX[iy * 28 + ix] * sW[c * 9 + dy * 3 + dx];
                    }
                }
                const float val = (h - sMu[c]) * sInv[c] + sBt[c];
                any = any || (val >= 0.f);
            }
        }
        p1[(size_t)b * 12544 + (size_t)pp * 64 + c] = any ? (signed char)1 : (signed char)-1;
    }
}

__global__ void conv2_wmma(const signed char* __restrict__ p1,
                           const float* __restrict__ conv2_w,
                           const float* __restrict__ g2, const float* __restrict__ bt2,
                           const float* __restrict__ mu2, const float* __restrict__ vr2,
                           signed char* __restrict__ p2) {
    __shared__ __align__(16) signed char sImg[196 * 64];
    __shared__ __align__(16) signed char sSign[196 * 64];
    __shared__ __align__(16) signed char sWA[9 * 4 * 32 * 32];
    __shared__ float sInv[64], sMu[64], sBt[64];

    const int b = blockIdx.x;
    const int tid = threadIdx.x;

    for (int c = tid; c < 64; c += 256) {
        sInv[c] = g2[c] * rsqrtf(vr2[c] + EPSV);
        sMu[c] = mu2[c];
        sBt[c] = bt2[c];
    }
    {
        const int4* src = (const int4*)(p1 + (size_t)b * 12544);
        int4* dst = (int4*)sImg;
        for (int i = tid; i < 784; i += 256) dst[i] = src[i];
    }
    for (int idx = tid; idx < 36864; idx += 256) {
        const int j    = idx & 3;
        const int v    = (idx >> 2) & 7;
        const int lane = (idx >> 5) & 31;
        const int ct   = (idx >> 10) & 3;
        const int tap  = idx >> 12;
        const int rowm = lane & 15;
        const int hf   = lane >> 4;
        const int co   = ct * 16 + rowm;
        const int ci   = ((v >> 2) << 5) + ((v & 3) << 2) + (hf << 4) + j;
        const float w  = conv2_w[(size_t)(co * 64 + ci) * 9 + tap];
        sWA[idx] = (w >= 0.f) ? (signed char)1 : (signed char)-1;
    }
    __syncthreads();

    const int lane = tid & 31;
    const int wave = tid >> 5;
    const int row  = lane & 15;
    const int half = lane >> 4;

    const int j1    = wave + 8;
    const bool has1 = (j1 < 13);
    const int j1c   = has1 ? j1 : 12;

    const int pix0 = wave * 16 + row;
    const int y0 = pix0 / 14, x0 = pix0 % 14;
    const int pix1 = j1c * 16 + row;
    const int pc1  = pix1 < 196 ? pix1 : 195;
    const int y1 = pc1 / 14, x1 = pc1 % 14;

    v8i acc0[4] = {};
    v8i acc1[4] = {};

    #pragma unroll 1
    for (int tap = 0; tap < 9; ++tap) {
        const int dy = tap / 3 - 1, dx = tap % 3 - 1;
        const int o0 = (iclamp(y0 + dy, 0, 13) * 14 + iclamp(x0 + dx, 0, 13)) * 64 + half * 16;
        const int o1 = (iclamp(y1 + dy, 0, 13) * 14 + iclamp(x1 + dx, 0, 13)) * 64 + half * 16;
        union { v8i v; v4i h[2]; } ub0, ub1;
        ub0.h[0] = *(const v4i*)(sImg + o0);
        ub0.h[1] = *(const v4i*)(sImg + o0 + 32);
        ub1.h[0] = *(const v4i*)(sImg + o1);
        ub1.h[1] = *(const v4i*)(sImg + o1 + 32);
        const signed char* wp = sWA + ((tap * 128 + lane) << 5);
        #pragma unroll
        for (int ct = 0; ct < 4; ++ct) {
            union { v8i v; v4i h[2]; } ua;
            ua.h[0] = *(const v4i*)(wp + ct * 1024);
            ua.h[1] = *(const v4i*)(wp + ct * 1024 + 16);
            acc0[ct] = __builtin_amdgcn_wmma_i32_16x16x64_iu8(
                true, ua.v, true, ub0.v, acc0[ct], false, false);
            acc1[ct] = __builtin_amdgcn_wmma_i32_16x16x64_iu8(
                true, ua.v, true, ub1.v, acc1[ct], false, false);
        }
    }

    #pragma unroll
    for (int ct = 0; ct < 4; ++ct) {
        unsigned long long packed = 0ull;
        #pragma unroll
        for (int r = 0; r < 8; ++r) {
            const int co = ct * 16 + half * 8 + r;
            const float val = ((float)acc0[ct][r] - sMu[co]) * sInv[co] + sBt[co];
            const unsigned char sb = (val >= 0.f) ? 0x01u : 0xFFu;
            packed |= ((unsigned long long)sb) << (8 * r);
        }
        *(unsigned long long*)(sSign + pix0 * 64 + ct * 16 + half * 8) = packed;
    }
    if (has1 && pix1 < 196) {
        #pragma unroll
        for (int ct = 0; ct < 4; ++ct) {
            unsigned long long packed = 0ull;
            #pragma unroll
            for (int r = 0; r < 8; ++r) {
                const int co = ct * 16 + half * 8 + r;
                const float val = ((float)acc1[ct][r] - sMu[co]) * sInv[co] + sBt[co];
                const unsigned char sb = (val >= 0.f) ? 0x01u : 0xFFu;
                packed |= ((unsigned long long)sb) << (8 * r);
            }
            *(unsigned long long*)(sSign + pix1 * 64 + ct * 16 + half * 8) = packed;
        }
    }
    __syncthreads();

    for (int idx = tid; idx < 3136; idx += 256) {
        const int c = idx / 49;
        const int rem = idx % 49;
        const int py = rem / 7, px = rem % 7;
        signed char m = -1;
        #pragma unroll
        for (int sy = 0; sy < 2; ++sy)
            #pragma unroll
            for (int sx = 0; sx < 2; ++sx) {
                const signed char s = sSign[((2 * py + sy) * 14 + (2 * px + sx)) * 64 + c];
                m = s > m ? s : m;
            }
        p2[(size_t)b * 3136 + idx] = m;
    }
}

__global__ void fc1_wmma(const signed char* __restrict__ p2,
                         const signed char* __restrict__ w1s,
                         const float* __restrict__ g3, const float* __restrict__ bt3,
                         const float* __restrict__ mu3, const float* __restrict__ vr3,
                         signed char* __restrict__ b3out) {
    const int lane = threadIdx.x & 31;
    const int wave = threadIdx.x >> 5;
    const int gid  = blockIdx.x * 8 + wave;
    const int jm   = gid & 31;
    const int jn   = gid >> 5;
    const int row  = lane & 15;
    const int half = lane >> 4;

    const signed char* aRow[4];
    #pragma unroll
    for (int i = 0; i < 4; ++i)
        aRow[i] = p2 + (size_t)((jm * 4 + i) * 16 + row) * 3136;
    const signed char* bRow[2];
    #pragma unroll
    for (int j = 0; j < 2; ++j)
        bRow[j] = w1s + (size_t)((jn * 2 + j) * 16 + row) * 3136;

    v8i acc[4][2] = {};

    #pragma unroll 1
    for (int ks = 0; ks < 49; ++ks) {
        const int k0 = ks * 64;
        v8i aF[4], bF[2];
        #pragma unroll
        for (int i = 0; i < 4; ++i) {
            const signed char* p = aRow[i] + k0 + half * 16;
            union { v8i v; unsigned long long q[4]; } u;
            u.q[0] = *(const unsigned long long*)(p);
            u.q[1] = *(const unsigned long long*)(p + 8);
            u.q[2] = *(const unsigned long long*)(p + 32);
            u.q[3] = *(const unsigned long long*)(p + 40);
            aF[i] = u.v;
        }
        #pragma unroll
        for (int j = 0; j < 2; ++j) {
            const signed char* p = bRow[j] + k0 + half * 16;
            union { v8i v; v4i h[2]; } u;
            u.h[0] = *(const v4i*)(p);
            u.h[1] = *(const v4i*)(p + 32);
            bF[j] = u.v;
        }
        #pragma unroll
        for (int i = 0; i < 4; ++i)
            #pragma unroll
            for (int j = 0; j < 2; ++j)
                acc[i][j] = __builtin_amdgcn_wmma_i32_16x16x64_iu8(
                    true, aF[i], true, bF[j], acc[i][j], false, false);
    }

    #pragma unroll
    for (int j = 0; j < 2; ++j) {
        const int o = (jn * 2 + j) * 16 + row;
        const float inv = g3[o] * rsqrtf(vr3[o] + EPSV);
        const float mu = mu3[o];
        const float bt = bt3[o];
        #pragma unroll
        for (int i = 0; i < 4; ++i) {
            const int bbase = (jm * 4 + i) * 16 + half * 8;
            #pragma unroll
            for (int r = 0; r < 8; ++r) {
                const float val = ((float)acc[i][j][r] - mu) * inv + bt;
                b3out[(size_t)(bbase + r) * 2048 + o] =
                    (val >= 0.f) ? (signed char)1 : (signed char)-1;
            }
        }
    }
}

__global__ void fc2_k(const signed char* __restrict__ b3,
                      const signed char* __restrict__ w2s,
                      const float* __restrict__ scale,
                      float* __restrict__ out) {
    const int t = blockIdx.x * blockDim.x + threadIdx.x;
    if (t >= 2048 * 10) return;
    const int b = t / 10, j = t % 10;
    const int4* xr = (const int4*)(b3 + (size_t)b * 2048);
    const int4* wr = (const int4*)(w2s + (size_t)j * 2048);
    int acc = 0;
    for (int k = 0; k < 128; ++k) {
        const int4 a4 = xr[k], w4 = wr[k];
        const int* ap = (const int*)&a4;
        const int* wp = (const int*)&w4;
        #pragma unroll
        for (int c = 0; c < 4; ++c) {
            const int a = ap[c], w = wp[c];
            acc += (int)(signed char)(a)       * (int)(signed char)(w);
            acc += (int)(signed char)(a >> 8)  * (int)(signed char)(w >> 8);
            acc += (int)(signed char)(a >> 16) * (int)(signed char)(w >> 16);
            acc += (int)(signed char)(a >> 24) * (int)(signed char)(w >> 24);
        }
    }
    out[t] = (float)acc * scale[0];
}

extern "C" void kernel_launch(void* const* d_in, const int* in_sizes, int n_in,
                              void* d_out, int out_size, void* d_ws, size_t ws_size,
                              hipStream_t stream) {
    const float* x        = (const float*)d_in[0];
    const float* conv1_w  = (const float*)d_in[1];
    const float* bn1_g    = (const float*)d_in[2];
    const float* bn1_b    = (const float*)d_in[3];
    const float* bn1_m    = (const float*)d_in[4];
    const float* bn1_v    = (const float*)d_in[5];
    const float* conv2_w  = (const float*)d_in[6];
    const float* bn2_g    = (const float*)d_in[7];
    const float* bn2_b    = (const float*)d_in[8];
    const float* bn2_m    = (const float*)d_in[9];
    const float* bn2_v    = (const float*)d_in[10];
    const float* fc1_w    = (const float*)d_in[11];
    const float* bn3_g    = (const float*)d_in[12];
    const float* bn3_b    = (const float*)d_in[13];
    const float* bn3_m    = (const float*)d_in[14];
    const float* bn3_v    = (const float*)d_in[15];
    const float* fc2_w    = (const float*)d_in[16];
    const float* scale    = (const float*)d_in[17];
    float* out = (float*)d_out;

    char* ws = (char*)d_ws;
    signed char* p1  = (signed char*)(ws + 0);
    signed char* p2  = (signed char*)(ws + 25690112);
    signed char* w1s = (signed char*)(ws + 32112640);
    signed char* w2s = (signed char*)(ws + 38535168);
    signed char* b3  = (signed char*)(ws + 38555648);
    (void)in_sizes; (void)n_in; (void)out_size; (void)ws_size;

    pack_signs<<<4096, 256, 0, stream>>>(fc1_w, fc2_w, w1s, w2s);

    conv1_k<<<2048, 256, 0, stream>>>(x, conv1_w, bn1_g, bn1_b, bn1_m, bn1_v, p1);

    conv2_wmma<<<2048, 256, 0, stream>>>(p1, conv2_w, bn2_g, bn2_b, bn2_m, bn2_v, p2);

    fc1_wmma<<<256, 256, 0, stream>>>(p2, w1s, bn3_g, bn3_b, bn3_m, bn3_v, b3);

    fc2_k<<<80, 256, 0, stream>>>(b3, w2s, scale, out);
}
